// NopadBloomAttention_28157805592831
// MI455X (gfx1250) — matrix operand order repaired and checked by reading, not compiled, not run on hardware
//
#include <hip/hip_runtime.h>
#include <hip/hip_bf16.h>
#include <math.h>

typedef __bf16 bf16_t;
typedef bf16_t v16bf __attribute__((ext_vector_type(16)));
typedef float  v8f   __attribute__((ext_vector_type(8)));
typedef unsigned short u16;
typedef unsigned int   u32;
typedef unsigned long long u64;

#define B_  4
#define S_  1024
#define H_  4096
#define NH_ 32
#define HD_ 128
#define T_  4096

__device__ __forceinline__ u16 f2bf(float f) {
    u32 x = __float_as_uint(f);
    x += 0x7fffu + ((x >> 16) & 1u);   // round-to-nearest-even
    return (u16)(x >> 16);
}

union FragAB { v16bf v; u32 u[8]; };

// 16-bit A-matrix (16x32 MxK) lane layout: VGPR i holds K pair starting at:
//   i<4 : 2i + half*8 ;  i>=4 : 16 + 2(i-4) + half*8
__device__ __forceinline__ int a_kpair(int i, int half) {
    int e = 2 * i;
    return (e & 7) + ((e >> 3) << 4) + half * 8;
}
// 16-bit B-matrix (32x16 KxN) lane layout: same K pairs as A-matrix
__device__ __forceinline__ int b_kpair(int i, int half) {
    int e = 2 * i;
    return (e & 7) + ((e >> 3) << 4) + half * 8;
}

// Per-lane async global->LDS copy of 16 bytes (GLOBAL_LOAD_ASYNC_TO_LDS_B128,
// ASYNCcnt-tracked).  lds_addr is the wave-relative LDS byte address (low 32
// bits of the generic pointer, per the ISA flat->LDS truncation rule).
__device__ __forceinline__ void async_copy_b128(const void* gptr, void* lptr) {
    u32 la = (u32)(u64)lptr;
    u64 ga = (u64)gptr;
    asm volatile("global_load_async_to_lds_b128 %0, %1, off"
                 :: "v"(la), "v"(ga) : "memory");
}
__device__ __forceinline__ void async_wait0() {
    asm volatile("s_wait_asynccnt 0x0" ::: "memory");
}

// ---------------------------------------------------------------------------
// GEMM: C[M=4096, N=4096] = A[4096,4096] x B[4096,4096]
// MODE 0: A fp32 (hidden), out -> bf16 q/k buffer, [b,nh,s,d] layout
// MODE 2: A fp32 (hidden), out -> bf16 v buffer,  [b,nh,d,s] (transposed)
// MODE 1: A bf16 (attn),   out -> fp32 d_out row-major
// ---------------------------------------------------------------------------
#define BM 128
#define BN 128
#define BK 64
#define LDA_S 72   // ushorts per As row (144B: 16B aligned, conflict-free frags)
#define LDB_S 72   // ushorts per Bs row, Bs is transposed [n][k]

template <int MODE>
__global__ __launch_bounds__(256)
void gemm_bf16_kernel(const float* __restrict__ Af32,
                      const u16*  __restrict__ Abf,
                      const float* __restrict__ Bw,
                      u16* __restrict__ out_bf,
                      float* __restrict__ out_f)
{
    __shared__ u16 As[BM * LDA_S];
    __shared__ u16 Bs[BN * LDB_S];

    const int tid  = threadIdx.x;
    const int lane = tid & 31;
    const int wid  = tid >> 5;
    const int half = lane >> 4;
    const int ln   = lane & 15;
    const int wm   = wid >> 2;      // 0..1 -> 64-row slab
    const int wn   = wid & 3;       // 0..3 -> 32-col slab

    const int m0 = blockIdx.y * BM;
    const int n0 = blockIdx.x * BN;

    v8f acc[4][2] = {};

    for (int k0 = 0; k0 < H_; k0 += BK) {
        // ---- stage A tile (BM x BK) into LDS as bf16, [m][k] ----
        if constexpr (MODE != 1) {
            #pragma unroll
            for (int i = 0; i < 8; ++i) {
                int c    = tid + 256 * i;      // 2048 float4 chunks
                int row  = c >> 4;             // 16 chunks per 64-float row
                int colc = (c & 15) << 2;
                float4 v = *(const float4*)(Af32 + (size_t)(m0 + row) * H_ + k0 + colc);
                ushort4 s;
                s.x = f2bf(v.x); s.y = f2bf(v.y); s.z = f2bf(v.z); s.w = f2bf(v.w);
                *(ushort4*)&As[row * LDA_S + colc] = s;
            }
        } else {
            // A already bf16: async DMA straight into LDS, no VGPR round-trip
            #pragma unroll
            for (int i = 0; i < 4; ++i) {
                int c    = tid + 256 * i;      // 1024 chunks of 8 ushorts
                int row  = c >> 3;
                int colc = (c & 7) << 3;
                async_copy_b128(Abf + (size_t)(m0 + row) * H_ + k0 + colc,
                                &As[row * LDA_S + colc]);
            }
        }
        // ---- stage B tile (BK x BN) transposed into LDS: Bs[n][k] ----
        #pragma unroll
        for (int i = 0; i < 8; ++i) {
            int c    = tid + 256 * i;          // 2048 float4 chunks
            int row  = c >> 5;                 // k index
            int colc = (c & 31) << 2;          // n index
            float4 v = *(const float4*)(Bw + (size_t)(k0 + row) * H_ + n0 + colc);
            Bs[(colc + 0) * LDB_S + row] = f2bf(v.x);
            Bs[(colc + 1) * LDB_S + row] = f2bf(v.y);
            Bs[(colc + 2) * LDB_S + row] = f2bf(v.z);
            Bs[(colc + 3) * LDB_S + row] = f2bf(v.w);
        }
        // prefetch next B tile into cache hierarchy (global_prefetch_b8)
        if (k0 + BK < H_) {
            __builtin_prefetch((const void*)(Bw + (size_t)(k0 + BK + (tid >> 3)) * H_
                                             + n0 + (tid & 7) * 16), 0, 1);
        }
        if constexpr (MODE == 1) async_wait0();
        __syncthreads();

        #pragma unroll
        for (int ks = 0; ks < 2; ++ks) {
            FragAB a[4], b[2];
            #pragma unroll
            for (int mf = 0; mf < 4; ++mf) {
                int m = wm * 64 + mf * 16 + ln;
                #pragma unroll
                for (int i = 0; i < 8; ++i) {
                    int k = a_kpair(i, half) + ks * 32;
                    a[mf].u[i] = *(const u32*)&As[m * LDA_S + k];
                }
            }
            #pragma unroll
            for (int nf = 0; nf < 2; ++nf) {
                int n = wn * 32 + nf * 16 + ln;
                #pragma unroll
                for (int i = 0; i < 8; ++i) {
                    int k = b_kpair(i, half) + ks * 32;
                    b[nf].u[i] = *(const u32*)&Bs[n * LDB_S + k];
                }
            }
            #pragma unroll
            for (int mf = 0; mf < 4; ++mf)
                #pragma unroll
                for (int nf = 0; nf < 2; ++nf)
                    acc[mf][nf] = __builtin_amdgcn_wmma_f32_16x16x32_bf16(
                        false, a[mf].v, false, b[nf].v, (short)0, acc[mf][nf],
                        false, false);
        }
        __syncthreads();
    }

    // ---- epilogue ----
    #pragma unroll
    for (int mf = 0; mf < 4; ++mf) {
        #pragma unroll
        for (int nf = 0; nf < 2; ++nf) {
            #pragma unroll
            for (int r = 0; r < 8; ++r) {
                int row = m0 + wm * 64 + mf * 16 + r + half * 8;
                int col = n0 + wn * 32 + nf * 16 + ln;
                float v = acc[mf][nf][r];
                int bb = row >> 10, s = row & 1023;      // t = bb*S + s
                int nh = col >> 7,  d = col & 127;       // e = nh*HD + d
                if constexpr (MODE == 0) {
                    out_bf[(((size_t)bb * NH_ + nh) * S_ + s) * HD_ + d] = f2bf(v);
                } else if constexpr (MODE == 2) {
                    // V stored transposed: [b][nh][d][s]
                    out_bf[(((size_t)bb * NH_ + nh) * HD_ + d) * S_ + s] = f2bf(v);
                } else {
                    out_f[(size_t)row * H_ + col] = v;
                }
            }
        }
    }
}

// ---------------------------------------------------------------------------
// Flash attention with ALiBi + causal mask.
// Block = 128 threads (4 wave32s); each wave owns 16 q-rows; kv tile = 32.
// K tile [kpos][d] and pre-transposed V tile [d][kpos] arrive via async DMA.
// ---------------------------------------------------------------------------
__global__ __launch_bounds__(128)
void attn_kernel(const u16* __restrict__ qbuf,
                 const u16* __restrict__ kbuf,
                 const u16* __restrict__ vbufT,   // [b][nh][d][s]
                 const int* __restrict__ seqlens,
                 u16* __restrict__ attn_bf)
{
    __shared__ u16 Ks[32 * 136];       // K tile  [kpos][d]
    __shared__ u16 Vt[128 * 40];       // V tile  [d][kpos]
    __shared__ u16 Ps[4 * 16 * 36];    // per-wave P scratch [m][kpos]

    const int tid  = threadIdx.x;
    const int lane = tid & 31;
    const int wv   = tid >> 5;
    const int half = lane >> 4;
    const int ln   = lane & 15;

    const int nh = blockIdx.y;
    const int b  = blockIdx.z;
    const int q0 = blockIdx.x * 64;
    const int qbase = q0 + wv * 16;
    const int seq = seqlens[b];

    const u16* Qp  = qbuf  + ((size_t)b * NH_ + nh) * S_ * HD_;
    const u16* Kp  = kbuf  + ((size_t)b * NH_ + nh) * S_ * HD_;
    const u16* VpT = vbufT + ((size_t)b * NH_ + nh) * HD_ * S_;

    const float sm_scale = 0.08838834764831845f;            // 1/sqrt(128)
    const float slope = exp2f(-0.25f * (float)(nh + 1));    // Bloom ALiBi, NH=32

    // Q A-fragments for d-steps 0..3 (K_wmma = head dim), loaded once
    FragAB qf[4];
    #pragma unroll
    for (int ds = 0; ds < 4; ++ds)
        #pragma unroll
        for (int i = 0; i < 8; ++i) {
            int d = a_kpair(i, half) + ds * 32;
            qf[ds].u[i] = *(const u32*)(Qp + (size_t)(qbase + ln) * HD_ + d);
        }

    v8f oacc[8] = {};
    float mst[8], lst[8];
    #pragma unroll
    for (int r = 0; r < 8; ++r) { mst[r] = -3.0e38f; lst[r] = 0.f; }

    int kend = q0 + 64; if (kend > seq) kend = seq;
    for (int kt = 0; kt < kend; kt += 32) {
        // ---- async K/V tile DMA (32 x 128 bf16 each, b128 per lane) ----
        #pragma unroll
        for (int i = 0; i < 4; ++i) {
            int c = tid + 128 * i;
            {   // K: 512 chunks, 16 chunks per kpos row
                int row = c >> 4, colc = (c & 15) << 3;
                async_copy_b128(Kp + (size_t)(kt + row) * HD_ + colc,
                                &Ks[row * 136 + colc]);
            }
            {   // V (transposed in global): 4 chunks per d row of 32 kpos
                int row = c >> 2, colc = (c & 3) << 3;
                async_copy_b128(VpT + (size_t)row * S_ + kt + colc,
                                &Vt[row * 40 + colc]);
            }
        }
        async_wait0();
        __syncthreads();

        // ---- scores: 16 q-rows x 32 k-cols, K_wmma = d (4 steps) ----
        v8f sc[2] = {};
        #pragma unroll
        for (int nf = 0; nf < 2; ++nf) {
            FragAB kf;
            #pragma unroll
            for (int ds = 0; ds < 4; ++ds) {
                #pragma unroll
                for (int i = 0; i < 8; ++i) {
                    int d = b_kpair(i, half) + ds * 32;
                    kf.u[i] = *(const u32*)&Ks[(nf * 16 + ln) * 136 + d];
                }
                sc[nf] = __builtin_amdgcn_wmma_f32_16x16x32_bf16(
                    false, qf[ds].v, false, kf.v, (short)0, sc[nf], false, false);
            }
        }

        // ---- ALiBi + causal mask + online softmax (per C-layout row r) ----
        #pragma unroll
        for (int r = 0; r < 8; ++r) {
            int qpos = qbase + r + half * 8;
            int k0p = kt + ln, k1p = kt + 16 + ln;
            float s0 = sc[0][r] * sm_scale + slope * (float)(k0p - qpos);
            float s1 = sc[1][r] * sm_scale + slope * (float)(k1p - qpos);
            if (k0p > qpos || k0p >= seq) s0 = -3.0e38f;
            if (k1p > qpos || k1p >= seq) s1 = -3.0e38f;
            float mx = fmaxf(s0, s1);
            #pragma unroll
            for (int m = 1; m < 16; m <<= 1) mx = fmaxf(mx, __shfl_xor(mx, m, 32));
            float mnew = fmaxf(mst[r], mx);
            float corr = __expf(mst[r] - mnew);
            float e0 = __expf(s0 - mnew);
            float e1 = __expf(s1 - mnew);
            float sum = e0 + e1;
            #pragma unroll
            for (int m = 1; m < 16; m <<= 1) sum += __shfl_xor(sum, m, 32);
            lst[r] = lst[r] * corr + sum;
            mst[r] = mnew;
            #pragma unroll
            for (int df = 0; df < 8; ++df) oacc[df][r] *= corr;
            // P to LDS (bf16), C-layout -> memory [m][kpos]
            Ps[(wv * 16 + r + half * 8) * 36 + ln]      = f2bf(e0);
            Ps[(wv * 16 + r + half * 8) * 36 + 16 + ln] = f2bf(e1);
        }
        asm volatile("s_wait_dscnt 0" ::: "memory");

        // ---- P.V : K_wmma = kpos (32), N = d (8 frags) ----
        FragAB pf;
        #pragma unroll
        for (int i = 0; i < 8; ++i)
            pf.u[i] = *(const u32*)&Ps[(wv * 16 + ln) * 36 + a_kpair(i, half)];
        #pragma unroll
        for (int df = 0; df < 8; ++df) {
            FragAB vf;
            #pragma unroll
            for (int i = 0; i < 8; ++i)
                vf.u[i] = *(const u32*)&Vt[(df * 16 + ln) * 40 + b_kpair(i, half)];
            oacc[df] = __builtin_amdgcn_wmma_f32_16x16x32_bf16(
                false, pf.v, false, vf.v, (short)0, oacc[df], false, false);
        }
        __syncthreads();
    }

    // ---- normalize and store attn output bf16 [t][e] ----
    #pragma unroll
    for (int df = 0; df < 8; ++df) {
        #pragma unroll
        for (int r = 0; r < 8; ++r) {
            int qpos = qbase + r + half * 8;
            float v = oacc[df][r] / lst[r];
            size_t t = (size_t)b * S_ + qpos;
            attn_bf[t * H_ + nh * HD_ + df * 16 + ln] = f2bf(v);
        }
    }
}

extern "C" void kernel_launch(void* const* d_in, const int* in_sizes, int n_in,
                              void* d_out, int out_size, void* d_ws, size_t ws_size,
                              hipStream_t stream)
{
    (void)in_sizes; (void)n_in; (void)out_size; (void)ws_size;
    const float* hidden = (const float*)d_in[0];
    const float* qkvw   = (const float*)d_in[1];
    const float* ow     = (const float*)d_in[2];
    const int*   seq    = (const int*)d_in[3];
    float* out = (float*)d_out;

    u16* qbuf = (u16*)d_ws;
    u16* kbuf = qbuf + (size_t)T_ * H_;
    u16* vbuf = kbuf + (size_t)T_ * H_;   // transposed [b][nh][d][s]
    u16* attn = vbuf + (size_t)T_ * H_;

    dim3 gg(H_ / BN, T_ / BM);
    gemm_bf16_kernel<0><<<gg, 256, 0, stream>>>(hidden, nullptr,
        qkvw + 0 * (size_t)H_ * H_, qbuf, nullptr);
    gemm_bf16_kernel<0><<<gg, 256, 0, stream>>>(hidden, nullptr,
        qkvw + 1 * (size_t)H_ * H_, kbuf, nullptr);
    gemm_bf16_kernel<2><<<gg, 256, 0, stream>>>(hidden, nullptr,
        qkvw + 2 * (size_t)H_ * H_, vbuf, nullptr);

    dim3 ga(S_ / 64, NH_, B_);
    attn_kernel<<<ga, 128, 0, stream>>>(qbuf, kbuf, vbuf, seq, attn);

    gemm_bf16_kernel<1><<<gg, 256, 0, stream>>>(nullptr, attn, ow, nullptr, out);
}
